// ResidualParallelWaveGANDiscriminator_54013508714721
// MI455X (gfx1250) — compile-verified, matrix operand order checked by reading, not run on hardware
//
#include <hip/hip_runtime.h>
#include <hip/hip_bf16.h>

// ---------------------------------------------------------------------------
// Problem constants (from reference)
// ---------------------------------------------------------------------------
#define BATCH   16
#define SEQL    32768
#define RCH     64      // residual channels
#define GCH     128     // gate channels (conv output)
#define KTAPS   3
#define LAYERS  30
#define TTILE   128     // timesteps per workgroup
#define NTILES  (TTILE / 16)
#define PAD     512     // max dilation -> zero guard on each side of h
#define HSTRIDE (SEQL + 2 * PAD)

typedef __bf16 bf16_t;
typedef __attribute__((ext_vector_type(16))) __bf16 v16bf;
typedef __attribute__((ext_vector_type(8)))  __bf16 v8bf;
typedef __attribute__((ext_vector_type(8)))  float  v8f;

// Packed-weight regions (element offsets into the bf16 wpack buffer).
// Each "fragment" is 512 bf16 = one 16x32 A-tile in WMMA lane order.
#define WC_OFF  0
#define WC_SZ   (LAYERS * 8 * 2 * KTAPS * 512)   // 737280
#define WO_OFF  (WC_OFF + WC_SZ)                 // 737280
#define WO_SZ   (LAYERS * 4 * 2 * 512)           // 122880
#define WS_OFF  (WO_OFF + WO_SZ)                 // 860160
#define WS_SZ   (LAYERS * 4 * 2 * 512)           // 122880
#define L1_OFF  (WS_OFF + WS_SZ)                 // 983040
#define L1_SZ   (4 * 2 * 512)                    // 4096
#define WPACK_TOTAL (L1_OFF + L1_SZ)             // 987136 bf16 elements

__device__ __forceinline__ v16bf cat8(v8bf lo, v8bf hi) {
  return __builtin_shufflevector(lo, hi, 0,1,2,3,4,5,6,7,8,9,10,11,12,13,14,15);
}

__device__ __forceinline__ float lrelu(float v) {
  return v >= 0.0f ? v : 0.2f * v;
}

__device__ __forceinline__ size_t hidx(int b, int t) {  // padded h indexing
  return ((size_t)b * HSTRIDE + PAD + t) * RCH;
}

// ---------------------------------------------------------------------------
// Weight packing: fp32 -> bf16 in WMMA 16x32 A-fragment lane order.
// A-matrix 16-bit layout (ISA 7.12.2): lane L<16 holds M=L, K in {0..7,16..23};
// lane L>=16 holds M=L-16, K in {8..15,24..31}; element e maps to
// klocal = (e>>3)*16 + (L>>4)*8 + ((e>>1)&3)*2 + (e&1).
// ---------------------------------------------------------------------------
__global__ void pack_weights_kernel(const float* __restrict__ Wc,
                                    const float* __restrict__ Ws,
                                    const float* __restrict__ Wo,
                                    const float* __restrict__ w_l1,
                                    bf16_t* __restrict__ wpack) {
  int idx = blockIdx.x * 256 + threadIdx.x;
  if (idx >= WPACK_TOTAL) return;
  int r    = idx & 511;
  int frag = idx >> 9;
  int lane = r >> 4;
  int e    = r & 15;
  int mlocal = lane & 15;
  int klocal = ((e >> 3) << 4) + ((lane >> 4) << 3) + (((e >> 1) & 3) << 1) + (e & 1);
  float v;
  if (idx < WO_OFF) {                      // Wc: frag = ((layer*8+mt)*2+kt)*3+tap
    int f = frag;
    int tap = f % 3; f /= 3;
    int kt  = f & 1; f >>= 1;
    int mt  = f & 7; f >>= 3;
    int layer = f;
    int row = mt * 16 + mlocal;            // output channel (GCH)
    int cin = kt * 32 + klocal;            // input channel (RCH)
    v = Wc[(((size_t)layer * GCH + row) * RCH + cin) * KTAPS + tap];
  } else if (idx < WS_OFF) {               // Wo: frag = (layer*4+mt)*2+kt
    int f = frag - (WO_OFF >> 9);
    int kt = f & 1; f >>= 1;
    int mt = f & 3; f >>= 2;
    int layer = f;
    v = Wo[((size_t)layer * RCH + mt * 16 + mlocal) * 64 + kt * 32 + klocal];
  } else if (idx < L1_OFF) {               // Ws
    int f = frag - (WS_OFF >> 9);
    int kt = f & 1; f >>= 1;
    int mt = f & 3; f >>= 2;
    int layer = f;
    v = Ws[((size_t)layer * RCH + mt * 16 + mlocal) * 64 + kt * 32 + klocal];
  } else {                                 // w_l1: frag = mt*2+kt
    int f = frag - (L1_OFF >> 9);
    int kt = f & 1; f >>= 1;
    int mt = f & 3;
    v = w_l1[(size_t)(mt * 16 + mlocal) * 64 + kt * 32 + klocal];
  }
  wpack[idx] = (bf16_t)v;
}

// ---------------------------------------------------------------------------
// Zero the guard (padding) regions of both h ping-pong buffers.
// Guard zeros implement conv zero-padding with no bounds checks in hot loops.
// 8 bf16 per thread; total chunks = 2 bufs * BATCH * 2*PAD*RCH / 8 = 262144.
// ---------------------------------------------------------------------------
__global__ void zero_guard_kernel(bf16_t* __restrict__ hA,
                                  bf16_t* __restrict__ hB) {
  const int cpb = (2 * PAD * RCH) / 8;               // chunks per batch = 8192
  int chunk = blockIdx.x * 256 + threadIdx.x;
  int buf = chunk / (BATCH * cpb);
  int rem = chunk % (BATCH * cpb);
  int b   = rem / cpb;
  int off = rem % cpb;
  int back = off / (cpb / 2);                        // 0 = front pad, 1 = back pad
  int o    = off % (cpb / 2);                        // chunk within region
  size_t elem = ((size_t)b * HSTRIDE + (back ? (PAD + SEQL) : 0)) * RCH +
                (size_t)o * 8;
  bf16_t* h = buf ? hB : hA;
  v8bf z = {};
  *(v8bf*)(h + elem) = z;
}

// ---------------------------------------------------------------------------
// First layer: h = lrelu(w_first * x + b_first), zero skips.
// h stored [b][PAD + t][64] bf16 (time-major, channel-contiguous).
// ---------------------------------------------------------------------------
__global__ void first_kernel(const float* __restrict__ x,
                             const float* __restrict__ w_first,
                             const float* __restrict__ b_first,
                             bf16_t* __restrict__ h,
                             float* __restrict__ skips) {
  int s = blockIdx.x * 4 + (threadIdx.x >> 6);   // flat (b,t) sample
  int c = threadIdx.x & 63;
  int b = s / SEQL, t = s % SEQL;
  float v = lrelu(w_first[c] * x[s] + b_first[c]);
  h[hidx(b, t) + c]          = (bf16_t)v;
  skips[(size_t)s * RCH + c] = 0.0f;
}

// ---------------------------------------------------------------------------
// One fused WaveNet layer:
//   stage 0: cooperatively stage 3 tap-shifted h tiles into LDS (once/block)
//   stage 1: y = dilated_conv(h)            (8 waves x [16 out-ch tile], WMMA)
//   stage 2: g = tanh(ya+bc)*sigmoid(yb+bc) (VALU, bf16 -> LDS)
//   stage 3: res = Wo*g (waves 0-3), skip += Ws*g (waves 4-7), WMMA
// ---------------------------------------------------------------------------
__global__ __launch_bounds__(256) void layer_kernel(
    const bf16_t* __restrict__ hin, bf16_t* __restrict__ hout,
    float* __restrict__ skips, const bf16_t* __restrict__ wpack,
    const float* __restrict__ bc, const float* __restrict__ bs,
    const float* __restrict__ bo, int layer, int dil) {

  __shared__ __align__(16) bf16_t hstg[KTAPS][TTILE][RCH]; // 48 KB
  __shared__ __align__(16) bf16_t ybuf[TTILE][GCH];        // 32 KB
  __shared__ __align__(16) bf16_t gbuf[TTILE][RCH];        // 16 KB

  const int tid  = threadIdx.x;
  const int wave = tid >> 5;
  const int lane = tid & 31;
  const int lhi  = lane >> 4;    // 0/1: which K half this lane owns
  const int lcol = lane & 15;    // N (time) column within tile
  const int b    = blockIdx.x / (SEQL / TTILE);
  const int t0   = (blockIdx.x % (SEQL / TTILE)) * TTILE;

  // ---- A fragments for the conv (issued early, overlap with staging) ------
  const int mBase = wave * 16;
  v16bf afrag[6];
  {
    const bf16_t* wl = wpack + WC_OFF + (size_t)layer * (8 * 2 * KTAPS * 512);
    #pragma unroll
    for (int kt = 0; kt < 2; ++kt)
      #pragma unroll
      for (int tap = 0; tap < KTAPS; ++tap)
        afrag[kt * 3 + tap] =
            *(const v16bf*)(wl + ((size_t)(wave * 2 + kt) * 3 + tap) * 512 + lane * 16);
  }

  // ---- stage 0: stage 3 tap-shifted tiles (guard pad makes this branchless)
  {
    bf16_t* dst = &hstg[0][0][0];
    #pragma unroll
    for (int tap = 0; tap < KTAPS; ++tap) {
      const bf16_t* src = hin + hidx(b, t0 + (tap - 1) * dil);
      #pragma unroll
      for (int it = 0; it < (TTILE * RCH) / (256 * 8); ++it) {  // 4 iters
        int e = (tid + it * 256) * 8;
        *(v8bf*)(dst + tap * TTILE * RCH + e) = *(const v8bf*)(src + e);
      }
    }
  }
  __syncthreads();

  // ---- stage 1: dilated conv, wave -> 16 output channels ------------------
  for (int nt = 0; nt < NTILES; ++nt) {
    v8f acc = {};
    const int tt = nt * 16 + lcol;
    #pragma unroll
    for (int tap = 0; tap < KTAPS; ++tap) {
      #pragma unroll
      for (int kt = 0; kt < 2; ++kt) {
        const bf16_t* p = &hstg[tap][tt][kt * 32 + lhi * 8];
        acc = __builtin_amdgcn_wmma_f32_16x16x32_bf16(
            false, afrag[kt * 3 + tap], false,
            cat8(*(const v8bf*)p, *(const v8bf*)(p + 16)),
            (short)0, acc, false, false);
      }
    }
    // C layout: lane holds rows mBase + lhi*8 + r, column = lcol
    v8bf yv = {};
    #pragma unroll
    for (int r = 0; r < 8; ++r) yv[r] = (bf16_t)acc[r];
    *(v8bf*)&ybuf[tt][mBase + lhi * 8] = yv;
  }
  __syncthreads();

  // ---- stage 2: gated activation ------------------------------------------
  {
    const float* bcl = bc + (size_t)layer * GCH;
    #pragma unroll
    for (int i = 0; i < (TTILE * RCH) / 256; ++i) {
      int e = tid + i * 256;
      int t = e >> 6, c = e & 63;
      float xa = (float)ybuf[t][c]      + bcl[c];
      float xb = (float)ybuf[t][c + 64] + bcl[c + 64];
      float g  = tanhf(xa) * (1.0f / (1.0f + __expf(-xb)));
      gbuf[t][c] = (bf16_t)g;
    }
  }
  __syncthreads();

  // ---- stage 3: residual (waves 0-3) / skip (waves 4-7) GEMMs -------------
  const int half = wave >> 2;            // 0 -> Wo/res, 1 -> Ws/skip
  const int m2   = (wave & 3) * 16;
  const size_t wOff = (half ? WS_OFF : WO_OFF) +
                      (size_t)layer * (4 * 2 * 512) + (size_t)(wave & 3) * 2 * 512;
  v16bf a2[2];
  a2[0] = *(const v16bf*)(wpack + wOff + 0 * 512 + lane * 16);
  a2[1] = *(const v16bf*)(wpack + wOff + 1 * 512 + lane * 16);
  const float* brow = (half ? bs : bo) + (size_t)layer * RCH + m2 + lhi * 8;
  float bias[8];
  #pragma unroll
  for (int r = 0; r < 8; ++r) bias[r] = brow[r];

  for (int nt = 0; nt < NTILES; ++nt) {
    v8f acc = {};
    const int tt = nt * 16 + lcol;
    #pragma unroll
    for (int kt = 0; kt < 2; ++kt) {
      const bf16_t* p = &gbuf[tt][kt * 32 + lhi * 8];
      acc = __builtin_amdgcn_wmma_f32_16x16x32_bf16(
          false, a2[kt], false, cat8(*(const v8bf*)p, *(const v8bf*)(p + 16)),
          (short)0, acc, false, false);
    }
    const int t = t0 + tt;
    if (half == 0) {
      const size_t base = hidx(b, t) + m2 + lhi * 8;
      v8bf hold = *(const v8bf*)(hin + base);
      v8bf hnew = {};
      #pragma unroll
      for (int r = 0; r < 8; ++r)
        hnew[r] = (bf16_t)((acc[r] + bias[r] + (float)hold[r]) * 0.70710678118654752f);
      *(v8bf*)(hout + base) = hnew;
    } else {
      float* sp = skips + ((size_t)b * SEQL + t) * RCH + m2 + lhi * 8;
      #pragma unroll
      for (int r = 0; r < 8; ++r) sp[r] += acc[r] + bias[r];
    }
  }
}

// ---------------------------------------------------------------------------
// Tail: out = w_l2 . lrelu(w_l1 . lrelu(skips*scale) + b_l1) + b_l2
// ---------------------------------------------------------------------------
__global__ __launch_bounds__(256) void final_kernel(
    const float* __restrict__ skips, const bf16_t* __restrict__ wpack,
    const float* __restrict__ b_l1, const float* __restrict__ w_l2,
    const float* __restrict__ b_l2, float* __restrict__ out) {

  __shared__ __align__(16) bf16_t sbuf[TTILE][RCH];
  __shared__ __align__(16) bf16_t zbuf[TTILE][RCH];

  const int tid  = threadIdx.x;
  const int wave = tid >> 5;
  const int lane = tid & 31;
  const int lhi  = lane >> 4;
  const int lcol = lane & 15;
  const int b    = blockIdx.x / (SEQL / TTILE);
  const int t0   = (blockIdx.x % (SEQL / TTILE)) * TTILE;
  const float SCALE = 0.18257418583505536f;   // sqrt(1/30)

  #pragma unroll
  for (int i = 0; i < (TTILE * RCH) / 256; ++i) {
    int e = tid + i * 256;
    int t = e >> 6, c = e & 63;
    float v = lrelu(skips[((size_t)b * SEQL + t0 + t) * RCH + c] * SCALE);
    sbuf[t][c] = (bf16_t)v;
  }
  __syncthreads();

  const int m2 = (wave & 3) * 16;
  v16bf a2[2];
  a2[0] = *(const v16bf*)(wpack + L1_OFF + ((size_t)(wave & 3) * 2 + 0) * 512 + lane * 16);
  a2[1] = *(const v16bf*)(wpack + L1_OFF + ((size_t)(wave & 3) * 2 + 1) * 512 + lane * 16);
  float bias[8];
  #pragma unroll
  for (int r = 0; r < 8; ++r) bias[r] = b_l1[m2 + lhi * 8 + r];

  const int ntBase = (wave >> 2) * 4;
  for (int nti = 0; nti < 4; ++nti) {
    const int tt = (ntBase + nti) * 16 + lcol;
    v8f acc = {};
    #pragma unroll
    for (int kt = 0; kt < 2; ++kt) {
      const bf16_t* p = &sbuf[tt][kt * 32 + lhi * 8];
      acc = __builtin_amdgcn_wmma_f32_16x16x32_bf16(
          false, a2[kt], false, cat8(*(const v8bf*)p, *(const v8bf*)(p + 16)),
          (short)0, acc, false, false);
    }
    #pragma unroll
    for (int r = 0; r < 8; ++r)
      zbuf[tt][m2 + lhi * 8 + r] = (bf16_t)lrelu(acc[r] + bias[r]);
  }
  __syncthreads();

  if (tid < TTILE) {
    float sum = b_l2[0];
    #pragma unroll 8
    for (int c = 0; c < RCH; ++c) sum += w_l2[c] * (float)zbuf[tid][c];
    out[(size_t)b * SEQL + t0 + tid] = sum;
  }
}

// ---------------------------------------------------------------------------
// Host launcher
// ---------------------------------------------------------------------------
extern "C" void kernel_launch(void* const* d_in, const int* in_sizes, int n_in,
                              void* d_out, int out_size, void* d_ws, size_t ws_size,
                              hipStream_t stream) {
  const float* x       = (const float*)d_in[0];
  const float* w_first = (const float*)d_in[1];
  const float* b_first = (const float*)d_in[2];
  const float* Wc      = (const float*)d_in[3];
  const float* bc      = (const float*)d_in[4];
  const float* Ws      = (const float*)d_in[5];
  const float* bs      = (const float*)d_in[6];
  const float* Wo      = (const float*)d_in[7];
  const float* bo      = (const float*)d_in[8];
  const float* w_l1    = (const float*)d_in[9];
  const float* b_l1    = (const float*)d_in[10];
  const float* w_l2    = (const float*)d_in[11];
  const float* b_l2    = (const float*)d_in[12];

  uint8_t* ws = (uint8_t*)d_ws;
  const size_t HBYTES  = (size_t)BATCH * HSTRIDE * RCH * sizeof(bf16_t); // ~66 MiB
  const size_t SKBYTES = (size_t)BATCH * SEQL * RCH * sizeof(float);    // 128 MiB
  bf16_t* hA    = (bf16_t*)(ws);
  bf16_t* hB    = (bf16_t*)(ws + HBYTES);
  float*  skips = (float*)(ws + 2 * HBYTES);
  bf16_t* wpack = (bf16_t*)(ws + 2 * HBYTES + SKBYTES);

  pack_weights_kernel<<<(WPACK_TOTAL + 255) / 256, 256, 0, stream>>>(
      Wc, Ws, Wo, w_l1, wpack);

  zero_guard_kernel<<<(2 * BATCH * 2 * PAD * RCH) / (8 * 256), 256, 0, stream>>>(
      hA, hB);

  first_kernel<<<(BATCH * SEQL) / 4, 256, 0, stream>>>(
      x, w_first, b_first, hA, skips);

  bf16_t* hin = hA;
  bf16_t* hout = hB;
  const int nblk = (BATCH * SEQL) / TTILE;   // 4096
  for (int i = 0; i < LAYERS; ++i) {
    int dil = 1 << (i % (LAYERS / 3));       // 2^(i % 10)
    layer_kernel<<<nblk, 256, 0, stream>>>(
        hin, hout, skips, wpack, bc, bs, bo, i, dil);
    bf16_t* tmp = hin; hin = hout; hout = tmp;
  }

  final_kernel<<<nblk, 256, 0, stream>>>(
      skips, wpack, b_l1, w_l2, b_l2, (float*)d_out);
}
